// TASTFHomo_67078799228986
// MI455X (gfx1250) — matrix operand order repaired and checked by reading, not compiled, not run on hardware
//
#include <hip/hip_runtime.h>
#include <hip/hip_bf16.h>

typedef _Float16 f16;
typedef __attribute__((ext_vector_type(16))) _Float16 v16h;
typedef __attribute__((ext_vector_type(8)))  _Float16 v8h;
typedef __attribute__((ext_vector_type(8)))  float    v8f;

namespace {
constexpr int  kB = 16, kT = 12, kN = 2000, kE = 8000;
constexpr int  kM    = kB * kN;          // 32000 batched nodes
constexpr int  kEtot = kB * kE + kM;     // 160000 edges incl. self loops
constexpr int  kD = 64, kFF = 128, kNH = 4, kL = 2, kHZ = 3;
constexpr long kR = (long)kT * kM;       // 384000 rows
constexpr unsigned kNegMax = 0x00800000u; // fenc(-FLT_MAX)
}

// ---------- float order-preserving encode for atomicMax ----------
__device__ __forceinline__ unsigned fenc(float f) {
  unsigned u = __float_as_uint(f);
  return (u >> 31) ? ~u : (u | 0x80000000u);
}
__device__ __forceinline__ float fdec(unsigned k) {
  return (k >> 31) ? __uint_as_float(k & 0x7fffffffu) : __uint_as_float(~k);
}
__device__ __forceinline__ float lrelu(float x) { return x > 0.f ? x : 0.2f * x; }

// x_seq is [B,T,N]; node m = b*N+n
__device__ __forceinline__ float xval(const float* x, int t, int node) {
  int b = node / kN, n = node - b * kN;
  return x[((size_t)b * kT + t) * kN + n];
}

// ---------- utility kernels ----------
__global__ void fill_u32(unsigned* p, unsigned v, size_t n) {
  size_t i = (size_t)blockIdx.x * blockDim.x + threadIdx.x;
  if (i < n) p[i] = v;
}
__global__ void cvt_f32_f16(const float* __restrict__ s, f16* __restrict__ d, size_t n) {
  size_t i = (size_t)blockIdx.x * blockDim.x + threadIdx.x;
  if (i < n) d[i] = (f16)s[i];
}
// W2 given [32,64] (in,out) -> store [64,32] = [Nout,K]
__global__ void cvtT_w2(const float* __restrict__ s, f16* __restrict__ d) {
  int i = blockIdx.x * blockDim.x + threadIdx.x;
  if (i >= 32 * 64) return;
  int r = i / 64, c = i - r * 64;
  d[c * 32 + r] = (f16)s[i];
}
__global__ void build_edges(const int* __restrict__ ei, int* __restrict__ src, int* __restrict__ dst) {
  int i = blockIdx.x * blockDim.x + threadIdx.x;
  if (i >= kEtot) return;
  int s, d;
  if (i < kB * kE) {
    int b = i / kE, j = i - b * kE;
    s = ei[j] + b * kN;
    d = ei[kE + j] + b * kN;
  } else {
    s = d = i - kB * kE;
  }
  src[i] = s; dst[i] = d;
}

// ---------- generic WMMA GEMM: out[R,Nout] = A[R,K](f16) x W[Nout,K](f16) + bias ----------
union U16 { v16h v; v8h h[2]; };

__global__ __launch_bounds__(256) void gemm_wmma(
    const f16* __restrict__ A, const f16* __restrict__ W,
    const float* __restrict__ bias,
    float* __restrict__ outF, f16* __restrict__ outH,
    int K, int Nout, int doRelu)
{
  int wave = threadIdx.x >> 5;
  int lane = threadIdx.x & 31;
  int hf   = lane >> 4;       // half-wave
  int rr   = lane & 15;
  long tileR = (long)blockIdx.x * 8 + wave;   // 16-row tile index
  int  nbase = blockIdx.y * 64;

  const f16* Arow = A + (size_t)(tileR * 16 + rr) * K;
  v8f acc0 = {}; v8f acc1 = {}; v8f acc2 = {}; v8f acc3 = {};
  int ksteps = K >> 5;
  for (int ks = 0; ks < ksteps; ++ks) {
    int k0 = ks * 32;
    // A fragment (16x32 f16): lanes 0-15 hold K 0-15, lanes 16-31 hold K 16-31
    U16 a;
    a.h[0] = *(const v8h*)(Arow + k0 + hf * 16);
    a.h[1] = *(const v8h*)(Arow + k0 + hf * 16 + 8);
    // B fragments (32x16 f16): lane = output column, half-wave splits K 0-15 / 16-31
    const f16* Wb = W + (size_t)(nbase + rr) * K + k0 + hf * 16;
    v16h b0 = *(const v16h*)(Wb);
    v16h b1 = *(const v16h*)(Wb + (size_t)16 * K);
    v16h b2 = *(const v16h*)(Wb + (size_t)32 * K);
    v16h b3 = *(const v16h*)(Wb + (size_t)48 * K);
    acc0 = __builtin_amdgcn_wmma_f32_16x16x32_f16(false, a.v, false, b0, (short)0, acc0, false, false);
    acc1 = __builtin_amdgcn_wmma_f32_16x16x32_f16(false, a.v, false, b1, (short)0, acc1, false, false);
    acc2 = __builtin_amdgcn_wmma_f32_16x16x32_f16(false, a.v, false, b2, (short)0, acc2, false, false);
    acc3 = __builtin_amdgcn_wmma_f32_16x16x32_f16(false, a.v, false, b3, (short)0, acc3, false, false);
  }
  float bv0 = bias ? bias[nbase + rr]      : 0.f;
  float bv1 = bias ? bias[nbase + 16 + rr] : 0.f;
  float bv2 = bias ? bias[nbase + 32 + rr] : 0.f;
  float bv3 = bias ? bias[nbase + 48 + rr] : 0.f;
#pragma unroll
  for (int g = 0; g < 8; ++g) {
    long row = tileR * 16 + g + hf * 8;   // C/D layout: lanes 16-31 hold M+8
    size_t base = (size_t)row * Nout + nbase + rr;
    float v0 = acc0[g] + bv0, v1 = acc1[g] + bv1, v2 = acc2[g] + bv2, v3 = acc3[g] + bv3;
    if (doRelu) { v0 = fmaxf(v0, 0.f); v1 = fmaxf(v1, 0.f); v2 = fmaxf(v2, 0.f); v3 = fmaxf(v3, 0.f); }
    if (outF) { outF[base] = v0; outF[base + 16] = v1; outF[base + 32] = v2; outF[base + 48] = v3; }
    if (outH) { outH[base] = (f16)v0; outH[base + 16] = (f16)v1; outH[base + 32] = (f16)v2; outH[base + 48] = (f16)v3; }
  }
}

// ---------- GAT1 (in_dim=1, algebraically collapsed) ----------
__global__ void gat1_edge_max(const float* __restrict__ x, const int* __restrict__ src,
                              const int* __restrict__ dst, const float* __restrict__ W1,
                              const float* __restrict__ as1, const float* __restrict__ ad1,
                              unsigned* __restrict__ mx)
{
  long i = (long)blockIdx.x * blockDim.x + threadIdx.x;
  if (i >= (long)kT * kEtot) return;
  int t = (int)(i / kEtot), e = (int)(i % kEtot);
  int s = src[e], d = dst[e];
  float xs = xval(x, t, s), xd = xval(x, t, d);
  long rd = (long)t * kM + d;
#pragma unroll
  for (int h = 0; h < 4; ++h) {
    float S = 0.f, D = 0.f;
#pragma unroll
    for (int c = 0; c < 8; ++c) { float w = W1[h * 8 + c]; S += w * as1[h * 8 + c]; D += w * ad1[h * 8 + c]; }
    float sc = lrelu(xs * S + xd * D);
    atomicMax(&mx[rd * 4 + h], fenc(sc));
  }
}
__global__ void gat1_edge_acc(const float* __restrict__ x, const int* __restrict__ src,
                              const int* __restrict__ dst, const float* __restrict__ W1,
                              const float* __restrict__ as1, const float* __restrict__ ad1,
                              const unsigned* __restrict__ mx, float* __restrict__ wsum,
                              float* __restrict__ wx)
{
  long i = (long)blockIdx.x * blockDim.x + threadIdx.x;
  if (i >= (long)kT * kEtot) return;
  int t = (int)(i / kEtot), e = (int)(i % kEtot);
  int s = src[e], d = dst[e];
  float xs = xval(x, t, s), xd = xval(x, t, d);
  long rd = (long)t * kM + d;
#pragma unroll
  for (int h = 0; h < 4; ++h) {
    float S = 0.f, D = 0.f;
#pragma unroll
    for (int c = 0; c < 8; ++c) { float w = W1[h * 8 + c]; S += w * as1[h * 8 + c]; D += w * ad1[h * 8 + c]; }
    float sc = lrelu(xs * S + xd * D);
    float w = __expf(sc - fdec(mx[rd * 4 + h]));
    atomicAdd(&wsum[rd * 4 + h], w);
    atomicAdd(&wx[rd * 4 + h], w * xs);
  }
}
// out1[r, h*8+c] = relu(z[r,h]*W1[h*8+c] + b1)
__global__ void gat1_node(const float* __restrict__ wx, const float* __restrict__ wsum,
                          const float* __restrict__ W1, const float* __restrict__ b1,
                          f16* __restrict__ out1)
{
  long r = (long)blockIdx.x * blockDim.x + threadIdx.x;
  if (r >= kR) return;
  float z[4];
#pragma unroll
  for (int h = 0; h < 4; ++h) z[h] = wx[r * 4 + h] / (wsum[r * 4 + h] + 1e-16f);
#pragma unroll
  for (int j = 0; j < 32; ++j)
    out1[r * 32 + j] = (f16)fmaxf(z[j >> 3] * W1[j] + b1[j], 0.f);
}

// ---------- GAT2 ----------
__global__ void gat2_scores(const f16* __restrict__ H, const float* __restrict__ as2,
                            const float* __restrict__ ad2, float* __restrict__ aS,
                            float* __restrict__ aD)
{
  long r = (long)blockIdx.x * blockDim.x + threadIdx.x;
  if (r >= kR) return;
#pragma unroll
  for (int h = 0; h < 4; ++h) {
    float s = 0.f, d = 0.f;
#pragma unroll
    for (int c = 0; c < 16; ++c) {
      float hv = (float)H[r * 64 + h * 16 + c];
      s += hv * as2[h * 16 + c];
      d += hv * ad2[h * 16 + c];
    }
    aS[r * 4 + h] = s; aD[r * 4 + h] = d;
  }
}
__global__ void gat2_edge_max(const int* __restrict__ src, const int* __restrict__ dst,
                              const float* __restrict__ aS, const float* __restrict__ aD,
                              unsigned* __restrict__ mx)
{
  long i = (long)blockIdx.x * blockDim.x + threadIdx.x;
  if (i >= (long)kT * kEtot) return;
  int t = (int)(i / kEtot), e = (int)(i % kEtot);
  long rs = (long)t * kM + src[e], rd = (long)t * kM + dst[e];
#pragma unroll
  for (int h = 0; h < 4; ++h)
    atomicMax(&mx[rd * 4 + h], fenc(lrelu(aS[rs * 4 + h] + aD[rd * 4 + h])));
}
__global__ void gat2_edge_acc(const int* __restrict__ src, const int* __restrict__ dst,
                              const float* __restrict__ aS, const float* __restrict__ aD,
                              const unsigned* __restrict__ mx, float* __restrict__ wsum,
                              const f16* __restrict__ H, float* __restrict__ acc)
{
  long i = (long)blockIdx.x * blockDim.x + threadIdx.x;
  if (i >= (long)kT * kEtot) return;
  int t = (int)(i / kEtot), e = (int)(i % kEtot);
  long rs = (long)t * kM + src[e], rd = (long)t * kM + dst[e];
#pragma unroll
  for (int h = 0; h < 4; ++h) {
    float sc = lrelu(aS[rs * 4 + h] + aD[rd * 4 + h]);
    float w = __expf(sc - fdec(mx[rd * 4 + h]));
    atomicAdd(&wsum[rd * 4 + h], w);
#pragma unroll
    for (int c = 0; c < 16; ++c)
      atomicAdd(&acc[rd * 64 + h * 16 + c], w * (float)H[rs * 64 + h * 16 + c]);
  }
}
// seq = relu(acc/sum + b2) + PE; write f32 master + f16 copy
__global__ void gat2_node_pe(const float* __restrict__ acc, const float* __restrict__ wsum,
                             const float* __restrict__ b2, float* __restrict__ seq,
                             f16* __restrict__ seq16)
{
  long r = (long)blockIdx.x * blockDim.x + threadIdx.x;
  if (r >= kR) return;
  int t = (int)(r / kM);
#pragma unroll
  for (int d = 0; d < 64; ++d) {
    int h = d >> 4;
    float v = acc[r * 64 + d] / (wsum[r * 4 + h] + 1e-16f) + b2[d];
    v = fmaxf(v, 0.f);
    float freq = __expf(-0.14391156f * (float)(2 * (d >> 1))); // ln(10000)/64
    float ang = (float)t * freq;
    v += (d & 1) ? __cosf(ang) : __sinf(ang);
    seq[r * 64 + d] = v;
    seq16[r * 64 + d] = (f16)v;
  }
}

// ---------- per-node temporal attention (T=12, hd=16, 4 heads) ----------
__global__ void attn_kernel(const f16* __restrict__ qkv, f16* __restrict__ O)
{
  int idx = blockIdx.x * blockDim.x + threadIdx.x;
  if (idx >= kM * kNH) return;
  int m = idx % kM, h = idx / kM;
  const float scale = 0.25f; // 1/sqrt(16)
  for (int t = 0; t < kT; ++t) {
    const f16* qp = qkv + ((size_t)t * kM + m) * 192 + h * 16;
    float q[16];
#pragma unroll
    for (int c = 0; c < 16; ++c) q[c] = (float)qp[c];
    float sc[kT], mx = -3.4e38f;
    for (int s = 0; s < kT; ++s) {
      const f16* kp = qkv + ((size_t)s * kM + m) * 192 + 64 + h * 16;
      float dot = 0.f;
#pragma unroll
      for (int c = 0; c < 16; ++c) dot += q[c] * (float)kp[c];
      dot *= scale;
      sc[s] = dot;
      mx = fmaxf(mx, dot);
    }
    float sum = 0.f;
    for (int s = 0; s < kT; ++s) { sc[s] = __expf(sc[s] - mx); sum += sc[s]; }
    float inv = 1.f / sum;
    float o[16];
#pragma unroll
    for (int c = 0; c < 16; ++c) o[c] = 0.f;
    for (int s = 0; s < kT; ++s) {
      const f16* vp = qkv + ((size_t)s * kM + m) * 192 + 128 + h * 16;
      float w = sc[s] * inv;
#pragma unroll
      for (int c = 0; c < 16; ++c) o[c] += w * (float)vp[c];
    }
    f16* op = O + ((size_t)t * kM + m) * 64 + h * 16;
#pragma unroll
    for (int c = 0; c < 16; ++c) op[c] = (f16)o[c];
  }
}

// ---------- residual + LayerNorm (one wave32 per row of 64) ----------
__global__ __launch_bounds__(256) void ln_kernel(float* __restrict__ seq,
                                                 const float* __restrict__ add,
                                                 const float* __restrict__ g,
                                                 const float* __restrict__ b,
                                                 f16* __restrict__ out16)
{
  int lane = threadIdx.x & 31;
  long row = (long)blockIdx.x * 8 + (threadIdx.x >> 5);
  if (row >= kR) return;
  size_t o = (size_t)row * 64;
  float x0 = seq[o + lane] + add[o + lane];
  float x1 = seq[o + 32 + lane] + add[o + 32 + lane];
  float s = x0 + x1;
#pragma unroll
  for (int off = 16; off > 0; off >>= 1) s += __shfl_xor(s, off, 32);
  float mean = s * (1.f / 64.f);
  float d0 = x0 - mean, d1 = x1 - mean;
  float v = d0 * d0 + d1 * d1;
#pragma unroll
  for (int off = 16; off > 0; off >>= 1) v += __shfl_xor(v, off, 32);
  float rstd = rsqrtf(v * (1.f / 64.f) + 1e-5f);
  float y0 = d0 * rstd * g[lane] + b[lane];
  float y1 = d1 * rstd * g[32 + lane] + b[32 + lane];
  seq[o + lane] = y0;         seq[o + 32 + lane] = y1;
  out16[o + lane] = (f16)y0;  out16[o + 32 + lane] = (f16)y1;
}

// ---------- prediction head ----------
__global__ void head_kernel(const float* __restrict__ seq, const float* __restrict__ Wh,
                            const float* __restrict__ bh, float* __restrict__ out)
{
  int m = blockIdx.x * blockDim.x + threadIdx.x;
  if (m >= kM) return;
  const float* x = seq + ((size_t)(kT - 1) * kM + m) * 64;
  int b = m / kN, n = m - b * kN;
#pragma unroll
  for (int hz = 0; hz < kHZ; ++hz) {
    float s = bh[hz];
    for (int c = 0; c < 64; ++c) s += x[c] * Wh[hz * 64 + c];
    out[((size_t)b * kHZ + hz) * kN + n] = s;
  }
}

// ==================================================================
extern "C" void kernel_launch(void* const* d_in, const int* in_sizes, int n_in,
                              void* d_out, int out_size, void* d_ws, size_t ws_size,
                              hipStream_t stream) {
  (void)in_sizes; (void)n_in; (void)out_size; (void)ws_size;
  const float* x     = (const float*)d_in[0];
  const int*   ei    = (const int*)d_in[1];
  const float* W1    = (const float*)d_in[2];
  const float* as1   = (const float*)d_in[3];
  const float* ad1   = (const float*)d_in[4];
  const float* b1    = (const float*)d_in[5];
  const float* W2    = (const float*)d_in[6];
  const float* as2   = (const float*)d_in[7];
  const float* ad2   = (const float*)d_in[8];
  const float* b2    = (const float*)d_in[9];
  const float* Wqkv  = (const float*)d_in[10];
  const float* bqkv  = (const float*)d_in[11];
  const float* Wo    = (const float*)d_in[12];
  const float* bo    = (const float*)d_in[13];
  const float* Wf1   = (const float*)d_in[14];
  const float* bf1   = (const float*)d_in[15];
  const float* Wf2   = (const float*)d_in[16];
  const float* bf2   = (const float*)d_in[17];
  const float* g1    = (const float*)d_in[18];
  const float* be1   = (const float*)d_in[19];
  const float* g2    = (const float*)d_in[20];
  const float* be2   = (const float*)d_in[21];
  const float* Wh    = (const float*)d_in[22];
  const float* bh    = (const float*)d_in[23];
  float* out = (float*)d_out;

  // ---- workspace bump allocator (aliased buffers) ----
  char* ws = (char*)d_ws;
  size_t off = 0;
  auto alloc = [&](size_t bytes) { size_t r = off; off = (off + bytes + 255) & ~(size_t)255; return r; };
  int* srcI = (int*)(ws + alloc((size_t)kEtot * 4));
  int* dstI = (int*)(ws + alloc((size_t)kEtot * 4));
  const size_t SLOT = (size_t)kT * kM * 4;                 // [T,M,4] f32
  float* GB = (float*)(ws + alloc(4 * SLOT * sizeof(float)));
  f16* w16 = (f16*)(ws + alloc((size_t)(kL*192*64 + kL*64*64 + kL*128*64 + kL*64*128 + 64*32) * sizeof(f16)));
  f16* Wqkv16 = w16;
  f16* Wo16   = Wqkv16 + kL * 192 * 64;
  f16* Wf116  = Wo16   + kL * 64 * 64;
  f16* Wf216  = Wf116  + kL * 128 * 64;
  f16* W2T16  = Wf216  + kL * 64 * 128;
  f16*   F16A = (f16*)(ws + alloc((size_t)kR * 64 * sizeof(f16)));    // H2 / seq16 / O (aliased)
  f16*   QF16 = (f16*)(ws + alloc((size_t)kR * 192 * sizeof(f16)));   // out1 / qkv / FFN1 (aliased)
  float* SEQ  = (float*)(ws + alloc((size_t)kR * 64 * sizeof(float)));
  float* ACC  = (float*)(ws + alloc((size_t)kR * 64 * sizeof(float))); // wout / gemm f32-out (aliased)
  f16* OUT1 = QF16;

  float* gb0 = GB;             // GAT1 mx  | GAT2 aS
  float* gb1 = GB + SLOT;      // GAT1 sum | GAT2 aD
  float* gb2 = GB + 2 * SLOT;  // GAT1 wx  | GAT2 mx
  float* gb3 = GB + 3 * SLOT;  //          | GAT2 sum

  const int TB = 256;
  auto blks = [](long n, int tb) { return (unsigned)((n + tb - 1) / tb); };

  // weights -> f16
  cvt_f32_f16<<<blks(kL*192*64, TB), TB, 0, stream>>>(Wqkv, Wqkv16, kL*192*64);
  cvt_f32_f16<<<blks(kL*64*64,  TB), TB, 0, stream>>>(Wo,   Wo16,   kL*64*64);
  cvt_f32_f16<<<blks(kL*128*64, TB), TB, 0, stream>>>(Wf1,  Wf116,  kL*128*64);
  cvt_f32_f16<<<blks(kL*64*128, TB), TB, 0, stream>>>(Wf2,  Wf216,  kL*64*128);
  cvtT_w2<<<blks(32*64, TB), TB, 0, stream>>>(W2, W2T16);
  build_edges<<<blks(kEtot, TB), TB, 0, stream>>>(ei, srcI, dstI);

  // ---- GAT1 ----
  fill_u32<<<blks(SLOT, TB), TB, 0, stream>>>((unsigned*)gb0, kNegMax, SLOT);
  fill_u32<<<blks(SLOT, TB), TB, 0, stream>>>((unsigned*)gb1, 0u, SLOT);
  fill_u32<<<blks(SLOT, TB), TB, 0, stream>>>((unsigned*)gb2, 0u, SLOT);
  long nTE = (long)kT * kEtot;
  gat1_edge_max<<<blks(nTE, TB), TB, 0, stream>>>(x, srcI, dstI, W1, as1, ad1, (unsigned*)gb0);
  gat1_edge_acc<<<blks(nTE, TB), TB, 0, stream>>>(x, srcI, dstI, W1, as1, ad1,
                                                  (const unsigned*)gb0, gb1, gb2);
  gat1_node<<<blks(kR, TB), TB, 0, stream>>>(gb2, gb1, W1, b1, OUT1);

  // ---- GAT2 projection via WMMA: H2[R,64] = out1[R,32] x W2T[64,32] ----
  dim3 gG((unsigned)(kR / 16 / 8), 1);
  gemm_wmma<<<gG, TB, 0, stream>>>(OUT1, W2T16, nullptr, nullptr, F16A, 32, 64, 0);
  gat2_scores<<<blks(kR, TB), TB, 0, stream>>>(F16A, as2, ad2, gb0, gb1);
  fill_u32<<<blks(SLOT, TB), TB, 0, stream>>>((unsigned*)gb2, kNegMax, SLOT);
  fill_u32<<<blks(SLOT, TB), TB, 0, stream>>>((unsigned*)gb3, 0u, SLOT);
  fill_u32<<<blks((size_t)kR * 64, TB), TB, 0, stream>>>((unsigned*)ACC, 0u, (size_t)kR * 64);
  gat2_edge_max<<<blks(nTE, TB), TB, 0, stream>>>(srcI, dstI, gb0, gb1, (unsigned*)gb2);
  gat2_edge_acc<<<blks(nTE, TB), TB, 0, stream>>>(srcI, dstI, gb0, gb1,
                                                  (const unsigned*)gb2, gb3, F16A, ACC);
  gat2_node_pe<<<blks(kR, TB), TB, 0, stream>>>(ACC, gb3, b2, SEQ, F16A);

  // ---- Transformer encoder ----
  for (int l = 0; l < kL; ++l) {
    dim3 gq((unsigned)(kR / 16 / 8), 3);
    gemm_wmma<<<gq, TB, 0, stream>>>(F16A, Wqkv16 + (size_t)l * 192 * 64, bqkv + l * 192,
                                     nullptr, QF16, 64, 192, 0);
    attn_kernel<<<blks((long)kM * kNH, TB), TB, 0, stream>>>(QF16, F16A);
    dim3 go((unsigned)(kR / 16 / 8), 1);
    gemm_wmma<<<go, TB, 0, stream>>>(F16A, Wo16 + (size_t)l * 64 * 64, bo + l * 64,
                                     ACC, nullptr, 64, 64, 0);
    ln_kernel<<<(unsigned)(kR / 8), TB, 0, stream>>>(SEQ, ACC, g1 + l * 64, be1 + l * 64, F16A);
    dim3 gf1((unsigned)(kR / 16 / 8), 2);
    gemm_wmma<<<gf1, TB, 0, stream>>>(F16A, Wf116 + (size_t)l * 128 * 64, bf1 + l * 128,
                                      nullptr, QF16, 64, 128, 1);
    gemm_wmma<<<go, TB, 0, stream>>>(QF16, Wf216 + (size_t)l * 64 * 128, bf2 + l * 64,
                                     ACC, nullptr, 128, 64, 0);
    ln_kernel<<<(unsigned)(kR / 8), TB, 0, stream>>>(SEQ, ACC, g2 + l * 64, be2 + l * 64, F16A);
  }

  head_kernel<<<blks(kM, TB), TB, 0, stream>>>(SEQ, Wh, bh, out);
}
